// SpikeDrivenTransformer_34875134443858
// MI455X (gfx1250) — matrix operand order repaired and checked by reading, not compiled, not run on hardware
//
#include <hip/hip_runtime.h>

typedef __attribute__((ext_vector_type(16))) _Float16 v16h;
typedef __attribute__((ext_vector_type(8)))  _Float16 v8h;
typedef __attribute__((ext_vector_type(8)))  float    v8f;

namespace {
constexpr int kT = 4, kB = 16, kC = 512, kN = 256;
constexpr float kEps = 1e-5f;
}

// ---- CDNA5 async global->LDS copy (16B per lane), tracked by ASYNCcnt ----
__device__ __forceinline__ void async_copy_b128(void* lds_dst, const void* gsrc) {
  unsigned l = (unsigned)(unsigned long long)lds_dst;  // low 32 bits = LDS offset
  asm volatile("global_load_async_to_lds_b128 %0, %1, off" ::"v"(l), "v"(gsrc)
               : "memory");
}
template <int N>
__device__ __forceinline__ void s_wait_asynccnt() {
  asm volatile("s_wait_asynccnt %0" ::"i"(N) : "memory");
}

// ---------------- weight split: w -> f16 hi + f16 lo ----------------
__global__ __launch_bounds__(256) void split_w_kernel(const float* __restrict__ w,
                                                      _Float16* __restrict__ hi,
                                                      _Float16* __restrict__ lo, int n) {
  int i = blockIdx.x * blockDim.x + threadIdx.x;
  if (i >= n) return;
  float f = w[i];
  _Float16 h = (_Float16)f;
  hi[i] = h;
  lo[i] = (_Float16)(f - (float)h);
}

// ---------------- BN fold ----------------
__global__ __launch_bounds__(256) void bn_prep_kernel(const float* __restrict__ g,
                                                      const float* __restrict__ b,
                                                      const float* __restrict__ m,
                                                      const float* __restrict__ v,
                                                      const float* __restrict__ bias,
                                                      float* __restrict__ inv,
                                                      float* __restrict__ shift, int n) {
  int i = blockIdx.x * blockDim.x + threadIdx.x;
  if (i >= n) return;
  float iv = g[i] * rsqrtf(v[i] + kEps);
  float sh = b[i] - m[i] * iv;
  if (bias) sh += bias[i] * iv;
  inv[i] = iv;
  shift[i] = sh;
}

// ---------------- LIF over T: x [t][b][c][n] f32 -> xs spikes [t][b][n][c] f16 ----------------
__global__ __launch_bounds__(256) void lif_x_kernel(const float* __restrict__ x,
                                                    _Float16* __restrict__ xs) {
  const int S = kB * kN * kC;
  int i = blockIdx.x * blockDim.x + threadIdx.x;  // (b, n, c), c fastest
  if (i >= S) return;
  int b = i >> 17;              // / (kN*kC)
  int n = (i >> 9) & (kN - 1);  // / kC % kN
  int c = i & (kC - 1);
  float vm = 0.f;
#pragma unroll
  for (int t = 0; t < kT; ++t) {
    float xv = x[(((size_t)t * kB + b) * kC + c) * kN + n];  // strided; L2-resident
    vm += (xv - vm) * 0.5f;               // tau = 2
    float s = (vm >= 1.0f) ? 1.f : 0.f;   // v_th = 1
    xs[(size_t)t * S + i] = (_Float16)s;  // coalesced
    vm *= (1.f - s);
  }
}

// ---------------- fused conv1x1 (WMMA, hi/lo) + BN + LIF(T) -> spikes [t][b][n][c] ----------------
// grid: x = N/64, y = C/128, z = B. Tile 128(o) x 64(n), K-step 32, double-buffered LDS,
// async global->LDS staging. 8 waves; each wave: 32x32 sub-tile (2x2 WMMA tiles).
__global__ __launch_bounds__(256) void qkv_gemm_lif_kernel(
    const _Float16* __restrict__ xs, const _Float16* __restrict__ whi,
    const _Float16* __restrict__ wlo, const float* __restrict__ bninv,
    const float* __restrict__ bnshift, _Float16* __restrict__ spikes) {
  const int nBase = blockIdx.x * 64;
  const int oBase = blockIdx.y * 128;
  const int b = blockIdx.z;

  __shared__ __align__(32) _Float16 sAh[2][128 * 32];
  __shared__ __align__(32) _Float16 sAl[2][128 * 32];
  __shared__ __align__(32) _Float16 sB[2][64 * 32];  // [n][k], direct copy

  const int tid = threadIdx.x;
  const int lane = tid & 31;
  const int wave = tid >> 5;
  const int mw = (wave >> 1) * 32;
  const int nw = (wave & 1) * 32;

  const int arow = tid >> 1;       // 0..127
  const int acg = (tid & 1) * 16;  // 0/16
  const int brow = tid >> 2;       // 0..63
  const int bkg = (tid & 3) * 8;   // 0..24

  const int fr = lane & 15;
  const int ak0 = (lane < 16) ? 0 : 16;
  const int bk0 = (lane < 16) ? 0 : 16;
  const int mAdd = (lane < 16) ? 0 : 8;

  const v8f vzero = {0.f, 0.f, 0.f, 0.f, 0.f, 0.f, 0.f, 0.f};
  v8f vmem[2][2];
#pragma unroll
  for (int mi = 0; mi < 2; ++mi)
#pragma unroll
    for (int ni = 0; ni < 2; ++ni) vmem[mi][ni] = vzero;

  for (int t = 0; t < kT; ++t) {
    v8f acc[2][2];
#pragma unroll
    for (int mi = 0; mi < 2; ++mi)
#pragma unroll
      for (int ni = 0; ni < 2; ++ni) acc[mi][ni] = vzero;

    const _Float16* xbase = xs + (((size_t)t * kB + b) * kN) * kC;  // [n][c]

    // 5 async b128 per wave per tile: Ah x2, Al x2, B x1
    auto stage = [&](int buf, int k0) {
      async_copy_b128(&sAh[buf][arow * 32 + acg],
                      whi + (size_t)(oBase + arow) * kC + k0 + acg);
      async_copy_b128(&sAh[buf][arow * 32 + acg + 8],
                      whi + (size_t)(oBase + arow) * kC + k0 + acg + 8);
      async_copy_b128(&sAl[buf][arow * 32 + acg],
                      wlo + (size_t)(oBase + arow) * kC + k0 + acg);
      async_copy_b128(&sAl[buf][arow * 32 + acg + 8],
                      wlo + (size_t)(oBase + arow) * kC + k0 + acg + 8);
      async_copy_b128(&sB[buf][brow * 32 + bkg],
                      xbase + (size_t)(nBase + brow) * kC + k0 + bkg);
    };

    stage(0, 0);
    int cur = 0;
    for (int k0 = 0; k0 < kC; k0 += 32) {
      const bool notLast = (k0 + 32 < kC);
      if (notLast) stage(cur ^ 1, k0 + 32);  // prefetch next tile
      if (notLast) s_wait_asynccnt<5>();     // current tile's 5 done (in-order)
      else         s_wait_asynccnt<0>();
      __syncthreads();

      v16h ah[2], al[2], bf[2];
#pragma unroll
      for (int mi = 0; mi < 2; ++mi) {
        const _Float16* ap = &sAh[cur][(mw + mi * 16 + fr) * 32 + ak0];
        v8h c0 = *(const v8h*)ap;
        v8h c1 = *(const v8h*)(ap + 8);
        ah[mi] = __builtin_shufflevector(c0, c1, 0, 1, 2, 3, 4, 5, 6, 7, 8, 9, 10,
                                         11, 12, 13, 14, 15);
        const _Float16* lp = &sAl[cur][(mw + mi * 16 + fr) * 32 + ak0];
        v8h d0 = *(const v8h*)lp;
        v8h d1 = *(const v8h*)(lp + 8);
        al[mi] = __builtin_shufflevector(d0, d1, 0, 1, 2, 3, 4, 5, 6, 7, 8, 9, 10,
                                         11, 12, 13, 14, 15);
      }
#pragma unroll
      for (int ni = 0; ni < 2; ++ni)
        bf[ni] = *(const v16h*)&sB[cur][(nw + ni * 16 + fr) * 32 + bk0];

#pragma unroll
      for (int mi = 0; mi < 2; ++mi)
#pragma unroll
        for (int ni = 0; ni < 2; ++ni) {
          acc[mi][ni] = __builtin_amdgcn_wmma_f32_16x16x32_f16(
              false, ah[mi], false, bf[ni], (short)0, acc[mi][ni], false, false);
          acc[mi][ni] = __builtin_amdgcn_wmma_f32_16x16x32_f16(
              false, al[mi], false, bf[ni], (short)0, acc[mi][ni], false, false);
        }
      __syncthreads();
      cur ^= 1;
    }

    // epilogue: BN + LIF; 8 consecutive o per (lane,tile) -> single b128 spike store
#pragma unroll
    for (int mi = 0; mi < 2; ++mi)
#pragma unroll
      for (int ni = 0; ni < 2; ++ni) {
        int nIdx = nBase + nw + ni * 16 + fr;
        int oB = oBase + mw + mi * 16 + mAdd;
        v8h sp;
#pragma unroll
        for (int r = 0; r < 8; ++r) {
          int o = oB + r;
          float u = acc[mi][ni][r] * bninv[o] + bnshift[o];
          float vm = vmem[mi][ni][r];
          vm += (u - vm) * 0.5f;
          float s = (vm >= 1.0f) ? 1.f : 0.f;
          sp[r] = (_Float16)s;
          vmem[mi][ni][r] = vm * (1.f - s);
        }
        *(v8h*)&spikes[(((size_t)t * kB + b) * kN + nIdx) * kC + oB] = sp;
      }
  }
}

// ---------------- kv = LIF_t( sum_n k*v ), v_th=0.5; [t][b][n][c] -> [t][b][c] ----------------
__global__ __launch_bounds__(256) void kv_lif_kernel(const _Float16* __restrict__ ks,
                                                     const _Float16* __restrict__ vs,
                                                     _Float16* __restrict__ kvs) {
  int i = blockIdx.x * blockDim.x + threadIdx.x;  // (b, c/8)
  if (i >= kB * kC / 8) return;
  int b = i / (kC / 8);
  int cg = (i % (kC / 8)) * 8;
  float vm[8];
#pragma unroll
  for (int j = 0; j < 8; ++j) vm[j] = 0.f;
  for (int t = 0; t < kT; ++t) {
    const _Float16* kp = ks + ((size_t)(t * kB + b) * kN) * kC + cg;
    const _Float16* vp = vs + ((size_t)(t * kB + b) * kN) * kC + cg;
    float s[8];
#pragma unroll
    for (int j = 0; j < 8; ++j) s[j] = 0.f;
    for (int n = 0; n < kN; ++n) {
      v8h a = *(const v8h*)(kp + (size_t)n * kC);  // coalesced across threads
      v8h c = *(const v8h*)(vp + (size_t)n * kC);
#pragma unroll
      for (int j = 0; j < 8; ++j) s[j] += (float)a[j] * (float)c[j];
    }
    v8h sp;
#pragma unroll
    for (int j = 0; j < 8; ++j) {
      vm[j] += (s[j] - vm[j]) * 0.5f;
      float p = (vm[j] >= 0.5f) ? 1.f : 0.f;
      sp[j] = (_Float16)p;
      vm[j] *= (1.f - p);
    }
    *(v8h*)&kvs[(size_t)(t * kB + b) * kC + cg] = sp;
  }
}

// ---------------- proj conv1x1 on attn = q*kv (binary) + bias/BN + residual ----------------
// grid: x = N/64, y = C/128, z = T*B. Same async double-buffered scheme; kv broadcast
// folded into the B fragment with packed f16 multiplies.
__global__ __launch_bounds__(256) void proj_gemm_kernel(
    const _Float16* __restrict__ qs, const _Float16* __restrict__ kvs,
    const _Float16* __restrict__ whi, const _Float16* __restrict__ wlo,
    const float* __restrict__ pinv, const float* __restrict__ pshift,
    const float* __restrict__ x, float* __restrict__ out) {
  const int nBase = blockIdx.x * 64;
  const int oBase = blockIdx.y * 128;
  const int tb = blockIdx.z;

  __shared__ __align__(32) _Float16 sAh[2][128 * 32];
  __shared__ __align__(32) _Float16 sAl[2][128 * 32];
  __shared__ __align__(32) _Float16 sB[2][64 * 32];

  const int tid = threadIdx.x;
  const int lane = tid & 31;
  const int wave = tid >> 5;
  const int mw = (wave >> 1) * 32;
  const int nw = (wave & 1) * 32;

  const int arow = tid >> 1;
  const int acg = (tid & 1) * 16;
  const int brow = tid >> 2;
  const int bkg = (tid & 3) * 8;

  const int fr = lane & 15;
  const int ak0 = (lane < 16) ? 0 : 16;
  const int bk0 = (lane < 16) ? 0 : 16;
  const int mAdd = (lane < 16) ? 0 : 8;

  const v8f vzero = {0.f, 0.f, 0.f, 0.f, 0.f, 0.f, 0.f, 0.f};
  v8f acc[2][2];
#pragma unroll
  for (int mi = 0; mi < 2; ++mi)
#pragma unroll
    for (int ni = 0; ni < 2; ++ni) acc[mi][ni] = vzero;

  const _Float16* qbase = qs + (size_t)tb * kN * kC;  // [n][c]
  const _Float16* kvbase = kvs + (size_t)tb * kC;

  auto stage = [&](int buf, int k0) {
    async_copy_b128(&sAh[buf][arow * 32 + acg],
                    whi + (size_t)(oBase + arow) * kC + k0 + acg);
    async_copy_b128(&sAh[buf][arow * 32 + acg + 8],
                    whi + (size_t)(oBase + arow) * kC + k0 + acg + 8);
    async_copy_b128(&sAl[buf][arow * 32 + acg],
                    wlo + (size_t)(oBase + arow) * kC + k0 + acg);
    async_copy_b128(&sAl[buf][arow * 32 + acg + 8],
                    wlo + (size_t)(oBase + arow) * kC + k0 + acg + 8);
    async_copy_b128(&sB[buf][brow * 32 + bkg],
                    qbase + (size_t)(nBase + brow) * kC + k0 + bkg);
  };

  stage(0, 0);
  int cur = 0;
  for (int k0 = 0; k0 < kC; k0 += 32) {
    const bool notLast = (k0 + 32 < kC);
    if (notLast) stage(cur ^ 1, k0 + 32);
    if (notLast) s_wait_asynccnt<5>();
    else         s_wait_asynccnt<0>();
    __syncthreads();

    // kv values for this lane's K chunk (half-wave-uniform 32B, L0-hot)
    v16h kvf = *(const v16h*)(kvbase + k0 + bk0);

    v16h ah[2], al[2], bf[2];
#pragma unroll
    for (int mi = 0; mi < 2; ++mi) {
      const _Float16* ap = &sAh[cur][(mw + mi * 16 + fr) * 32 + ak0];
      v8h c0 = *(const v8h*)ap;
      v8h c1 = *(const v8h*)(ap + 8);
      ah[mi] = __builtin_shufflevector(c0, c1, 0, 1, 2, 3, 4, 5, 6, 7, 8, 9, 10, 11,
                                       12, 13, 14, 15);
      const _Float16* lp = &sAl[cur][(mw + mi * 16 + fr) * 32 + ak0];
      v8h d0 = *(const v8h*)lp;
      v8h d1 = *(const v8h*)(lp + 8);
      al[mi] = __builtin_shufflevector(d0, d1, 0, 1, 2, 3, 4, 5, 6, 7, 8, 9, 10, 11,
                                       12, 13, 14, 15);
    }
#pragma unroll
    for (int ni = 0; ni < 2; ++ni) {
      v16h qf = *(const v16h*)&sB[cur][(nw + ni * 16 + fr) * 32 + bk0];
      bf[ni] = qf * kvf;  // attn = q * kv (exact: both binary)
    }

#pragma unroll
    for (int mi = 0; mi < 2; ++mi)
#pragma unroll
      for (int ni = 0; ni < 2; ++ni) {
        acc[mi][ni] = __builtin_amdgcn_wmma_f32_16x16x32_f16(
            false, ah[mi], false, bf[ni], (short)0, acc[mi][ni], false, false);
        acc[mi][ni] = __builtin_amdgcn_wmma_f32_16x16x32_f16(
            false, al[mi], false, bf[ni], (short)0, acc[mi][ni], false, false);
      }
    __syncthreads();
    cur ^= 1;
  }

#pragma unroll
  for (int mi = 0; mi < 2; ++mi)
#pragma unroll
    for (int ni = 0; ni < 2; ++ni) {
      int nIdx = nBase + nw + ni * 16 + fr;
#pragma unroll
      for (int r = 0; r < 8; ++r) {
        int o = oBase + mw + mi * 16 + r + mAdd;
        size_t oidx = ((size_t)tb * kC + o) * kN + nIdx;
        out[oidx] = acc[mi][ni][r] * pinv[o] + pshift[o] + x[oidx];
      }
    }
}

extern "C" void kernel_launch(void* const* d_in, const int* in_sizes, int n_in,
                              void* d_out, int out_size, void* d_ws, size_t ws_size,
                              hipStream_t stream) {
  (void)in_sizes; (void)n_in; (void)out_size; (void)ws_size;
  const float* x      = (const float*)d_in[0];
  const float* q_w    = (const float*)d_in[1];
  const float* q_g    = (const float*)d_in[2];
  const float* q_b    = (const float*)d_in[3];
  const float* q_m    = (const float*)d_in[4];
  const float* q_v    = (const float*)d_in[5];
  const float* k_w    = (const float*)d_in[6];
  const float* k_g    = (const float*)d_in[7];
  const float* k_b    = (const float*)d_in[8];
  const float* k_m    = (const float*)d_in[9];
  const float* k_v    = (const float*)d_in[10];
  const float* v_w    = (const float*)d_in[11];
  const float* v_g    = (const float*)d_in[12];
  const float* v_b    = (const float*)d_in[13];
  const float* v_m    = (const float*)d_in[14];
  const float* v_v    = (const float*)d_in[15];
  const float* proj_w = (const float*)d_in[16];
  const float* proj_b = (const float*)d_in[17];
  const float* p_g    = (const float*)d_in[18];
  const float* p_b2   = (const float*)d_in[19];
  const float* p_m    = (const float*)d_in[20];
  const float* p_v    = (const float*)d_in[21];

  char* ws = (char*)d_ws;
  size_t off = 0;
  auto bump = [&](size_t bytes) {
    size_t o = off;
    off += (bytes + 255) & ~(size_t)255;
    return o;
  };

  const size_t actBytes = (size_t)kT * kB * kC * kN * sizeof(_Float16);  // 16 MB
  _Float16* xs  = (_Float16*)(ws + bump(actBytes));
  _Float16* qsp = (_Float16*)(ws + bump(actBytes));
  _Float16* ksp = (_Float16*)(ws + bump(actBytes));
  _Float16* vsp = (_Float16*)(ws + bump(actBytes));
  _Float16* kvs = (_Float16*)(ws + bump((size_t)kT * kB * kC * sizeof(_Float16)));

  const size_t wElems = (size_t)kC * kC;
  _Float16 *wh[4], *wl[4];
  float *bni[4], *bns[4];
  for (int i = 0; i < 4; ++i) {
    wh[i] = (_Float16*)(ws + bump(wElems * sizeof(_Float16)));
    wl[i] = (_Float16*)(ws + bump(wElems * sizeof(_Float16)));
    bni[i] = (float*)(ws + bump(kC * sizeof(float)));
    bns[i] = (float*)(ws + bump(kC * sizeof(float)));
  }

  const int wn = (int)wElems;
  const int wBlocks = (wn + 255) / 256;
  split_w_kernel<<<wBlocks, 256, 0, stream>>>(q_w, wh[0], wl[0], wn);
  split_w_kernel<<<wBlocks, 256, 0, stream>>>(k_w, wh[1], wl[1], wn);
  split_w_kernel<<<wBlocks, 256, 0, stream>>>(v_w, wh[2], wl[2], wn);
  split_w_kernel<<<wBlocks, 256, 0, stream>>>(proj_w, wh[3], wl[3], wn);

  bn_prep_kernel<<<2, 256, 0, stream>>>(q_g, q_b, q_m, q_v, (const float*)nullptr, bni[0], bns[0], kC);
  bn_prep_kernel<<<2, 256, 0, stream>>>(k_g, k_b, k_m, k_v, (const float*)nullptr, bni[1], bns[1], kC);
  bn_prep_kernel<<<2, 256, 0, stream>>>(v_g, v_b, v_m, v_v, (const float*)nullptr, bni[2], bns[2], kC);
  bn_prep_kernel<<<2, 256, 0, stream>>>(p_g, p_b2, p_m, p_v, proj_b, bni[3], bns[3], kC);

  lif_x_kernel<<<(kB * kC * kN + 255) / 256, 256, 0, stream>>>(x, xs);

  dim3 gq(kN / 64, kC / 128, kB);
  qkv_gemm_lif_kernel<<<gq, 256, 0, stream>>>(xs, wh[0], wl[0], bni[0], bns[0], qsp);
  qkv_gemm_lif_kernel<<<gq, 256, 0, stream>>>(xs, wh[1], wl[1], bni[1], bns[1], ksp);
  qkv_gemm_lif_kernel<<<gq, 256, 0, stream>>>(xs, wh[2], wl[2], bni[2], bns[2], vsp);

  kv_lif_kernel<<<(kB * kC / 8 + 255) / 256, 256, 0, stream>>>(ksp, vsp, kvs);

  dim3 gp(kN / 64, kC / 128, kT * kB);
  proj_gemm_kernel<<<gp, 256, 0, stream>>>(qsp, kvs, wh[3], wl[3], bni[3], bns[3], x,
                                           (float*)d_out);
}
